// MultiHeadDeepAttention_32298154066191
// MI455X (gfx1250) — matrix operand order repaired and checked by reading, not compiled, not run on hardware
//
#include <hip/hip_runtime.h>

// ---------------------------------------------------------------------------
// MultiHeadDeepAttention for MI455X (gfx1250).
// B=2, S=2048, D=1024, H=16, head dim 64.
// Dense chains: TDM -> LDS (double buffered) -> ds_load fragments -> WMMA.
// Attention: flash-style, all matmuls on v_wmma_f32_16x16x32_bf16.
// ---------------------------------------------------------------------------

#define B_ 2
#define S_ 2048
#define D_ 1024
#define H_ 16
#define M_ (B_ * S_)      // 4096 token rows
#define FSTRIDE 6144      // concat-feature buffer stride
#define MAXFI 5120        // max single-layer input width

#define LDSROW 40         // ushorts per staged tile row: 32 data + 8 pad (80B)

typedef __attribute__((ext_vector_type(16))) __bf16        bf16x16;
typedef __attribute__((ext_vector_type(8)))  float         f32x8;
typedef __attribute__((ext_vector_type(4)))  unsigned int  u32x4;
typedef __attribute__((ext_vector_type(8)))  unsigned int  u32x8;
typedef __attribute__((ext_vector_type(4)))  int           i32x4;
typedef __attribute__((ext_vector_type(8)))  int           i32x8;

#if defined(__has_builtin)
#if __has_builtin(__builtin_amdgcn_tensor_load_to_lds) && \
    __has_builtin(__builtin_amdgcn_s_wait_tensorcnt)
#define USE_TDM 1
#endif
#endif
#ifndef USE_TDM
#define USE_TDM 0
#endif

__device__ __forceinline__ unsigned short f2bf(float f) {
  unsigned u = __builtin_bit_cast(unsigned, f);
  u += 0x7FFFu + ((u >> 16) & 1u);           // round-to-nearest-even
  return (unsigned short)(u >> 16);
}

// A-fragment (16x32 bf16, MxK): lane l holds row l&15;
// elements 0..15 = K 16*(l>>4)+0..15 contiguous.
__device__ __forceinline__ bf16x16 load_afrag(const unsigned short* p) {
  u32x8 w;
  *(u32x4*)&w       = *(const u32x4*)p;
  *((u32x4*)&w + 1) = *(const u32x4*)(p + 8);
  return __builtin_bit_cast(bf16x16, w);
}

// B-fragment (32x16 bf16, KxN): lane l holds col l&15, K = 16*(l>>4)+0..15
// contiguous -> 32B from a [N,K]-row-major (W^T) row.
__device__ __forceinline__ bf16x16 load_bfrag(const unsigned short* p) {
  u32x8 w;
  *(u32x4*)&w       = *(const u32x4*)p;
  *((u32x4*)&w + 1) = *(const u32x4*)(p + 8);
  return __builtin_bit_cast(bf16x16, w);
}

#if USE_TDM
// Issue one 2-D TDM tile load: tile_dim0 = 32 elems (K), tile_dim1 = rows,
// element size 2B, LDS padding 4 DWORDs every 16 DWORDs -> 80B row stride.
// D# encoding per CDNA5 ISA 8.3-8.4.
__device__ __forceinline__ void tdm_load_tile(unsigned lds_off,
                                              unsigned long long gaddr,
                                              int stride_elems, int rows) {
  u32x4 g0;
  g0[0] = 1u;                                              // count=1
  g0[1] = lds_off;                                         // lds_addr (bytes)
  g0[2] = (unsigned)gaddr;                                 // global_addr lo
  g0[3] = (unsigned)((gaddr >> 32) & 0x01FFFFFFull) | (2u << 30); // hi | type=2
  i32x8 g1;
  g1[0] = (1 << 16) | (1 << 20) | (3 << 22) | (3 << 25);   // 2B, pad 4dw/16dw
  g1[1] = 0;                                               // tdim0 = 1<<30 (lo16=0)
  g1[2] = 0x4000;                                          // tdim0 hi | tdim1 lo
  g1[3] = 0x4000 | (32 << 16);                             // tdim1 hi | tile0=32
  g1[4] = rows;                                            // tile1 | tile2=0
  g1[5] = stride_elems;                                    // dim0 stride lo32
  g1[6] = 0;
  g1[7] = 0;
  i32x4 gz = {0, 0, 0, 0};
#if __clang_major__ >= 23
  i32x8 gz8 = {0, 0, 0, 0, 0, 0, 0, 0};
  __builtin_amdgcn_tensor_load_to_lds(g0, g1, gz, gz, gz8, 0);
#else
  __builtin_amdgcn_tensor_load_to_lds(g0, g1, gz, gz, 0);
#endif
}
#else
// Fallback: cooperative stage of a 128x32 bf16 tile (row stride LDSROW).
__device__ __forceinline__ void stage_tile(unsigned short* dst,
                                           const unsigned short* src,
                                           int stride_elems, int tid) {
#pragma unroll
  for (int t = 0; t < 4; ++t) {
    int c = tid + t * 128;          // 512 chunks of 16B
    int row = c >> 2, part = c & 3;
    *(u32x4*)(dst + row * LDSROW + part * 8) =
        *(const u32x4*)(src + (size_t)row * stride_elems + part * 8);
  }
}
#endif

// ---------------------------------------------------------------------------
// feat-norm: per row, mean/var over fi features, affine, emit bf16 A matrix.
// ---------------------------------------------------------------------------
__global__ __launch_bounds__(256)
void featnorm_kernel(const float* __restrict__ X, int xstride, int fi,
                     const float* __restrict__ g, const float* __restrict__ bt,
                     unsigned short* __restrict__ out)
{
  __shared__ float red[256], red2[256];
  const int row = blockIdx.x;
  const float* x = X + (size_t)row * xstride;
  float s = 0.f, s2 = 0.f;
  for (int j = threadIdx.x; j < fi; j += 256) {
    float v = x[j]; s += v; s2 += v * v;
  }
  red[threadIdx.x] = s; red2[threadIdx.x] = s2;
  __syncthreads();
  for (int st = 128; st > 0; st >>= 1) {
    if ((int)threadIdx.x < st) {
      red[threadIdx.x]  += red[threadIdx.x + st];
      red2[threadIdx.x] += red2[threadIdx.x + st];
    }
    __syncthreads();
  }
  const float inv = 1.f / (float)fi;
  const float mu  = red[0] * inv;
  const float var = fmaxf(red2[0] * inv - mu * mu, 0.f);
  const float rs  = rsqrtf(var + 1e-5f);
  for (int j = threadIdx.x; j < fi; j += 256) {
    float v = (x[j] - mu) * rs * g[j] + bt[j];
    out[(size_t)row * fi + j] = f2bf(v);
  }
}

// W [K, 1024] f32  ->  W^T [1024, K] bf16
__global__ __launch_bounds__(256)
void wtconv_kernel(const float* __restrict__ W, unsigned short* __restrict__ WT,
                   int K)
{
  const int n = blockIdx.x;
  const int k = blockIdx.y * 256 + threadIdx.x;
  if (k < K) WT[(size_t)n * K + k] = f2bf(W[(size_t)k * D_ + n]);
}

// x [M,1024] f32 -> feats[:, 0:1024] (stride FSTRIDE)
__global__ __launch_bounds__(256)
void copyin_kernel(const float* __restrict__ x, float* __restrict__ feats)
{
  size_t i = (size_t)blockIdx.x * 256 + threadIdx.x;
  if (i < (size_t)M_ * D_) {
    size_t row = i >> 10, c = i & 1023;
    feats[row * FSTRIDE + c] = x[i];
  }
}

// V (bf16 [B,S,D]) -> V^T per head: VT[b][h][d(64)][s]
__global__ __launch_bounds__(256)
void vtrans_kernel(const unsigned short* __restrict__ Vb,
                   unsigned short* __restrict__ VT)
{
  size_t i = (size_t)blockIdx.x * 256 + threadIdx.x;
  if (i >= (size_t)B_ * H_ * 64 * S_) return;
  int    s  = (int)(i & (S_ - 1));
  size_t t  = i >> 11;
  int    dd = (int)(t & 63);
  size_t bh = t >> 6;
  int    h  = (int)(bh & (H_ - 1));
  int    b  = (int)(bh >> 4);
  VT[i] = Vb[((size_t)(b * S_ + s)) * D_ + h * 64 + dd];
}

// ---------------------------------------------------------------------------
// GEMM: out[M,1024] = relu(A[M,K](bf16) x W[K,1024] + bias)
// 128 thr = 4 waves; block tile 128x128; wave tile 64x64 (16 WMMA / k-step).
// A/B tiles staged to double-buffered LDS by the Tensor Data Mover; TDM of
// step k+1 overlaps WMMA of step k; tiles published via tensorcnt + barrier.
// ---------------------------------------------------------------------------
__global__ __launch_bounds__(128)
void gemm_fn_kernel(const unsigned short* __restrict__ A,
                    const unsigned short* __restrict__ WT,  // [1024, K]
                    const float* __restrict__ bias,
                    float* __restrict__ outf,
                    unsigned short* __restrict__ outb,
                    int K, int ostride, int ocol, int emit_bf16)
{
  __shared__ unsigned short ldsA[2][128 * LDSROW];
  __shared__ unsigned short ldsB[2][128 * LDSROW];

  const int tid  = threadIdx.x;
  const int lane = tid & 31;
  const int wave = tid >> 5;
  const int l15  = lane & 15;
  const int half = lane >> 4;
  const int m0   = blockIdx.x * 128;
  const int n0   = blockIdx.y * 128;
  const int rw   = (wave >> 1) * 64;   // wave row offset in block tile
  const int cw   = (wave & 1) * 64;    // wave col offset in block tile

  f32x8 acc[4][4];
#pragma unroll
  for (int i = 0; i < 4; ++i)
#pragma unroll
    for (int j = 0; j < 4; ++j)
#pragma unroll
      for (int r = 0; r < 8; ++r) acc[i][j][r] = 0.f;

  const int nk = K >> 5;

#if USE_TDM
  const unsigned long long Ag = (unsigned long long)A + ((size_t)m0 * K) * 2;
  const unsigned long long Bg = (unsigned long long)WT + ((size_t)n0 * K) * 2;
  if (wave == 0) {
    tdm_load_tile((unsigned)(unsigned long long)&ldsA[0][0], Ag, K, 128);
    tdm_load_tile((unsigned)(unsigned long long)&ldsB[0][0], Bg, K, 128);
    __builtin_amdgcn_s_wait_tensorcnt((short)0);
  }
#else
  stage_tile(&ldsA[0][0], A  + (size_t)m0 * K, K, tid);
  stage_tile(&ldsB[0][0], WT + (size_t)n0 * K, K, tid);
#endif
  __syncthreads();

  for (int ks = 0; ks < nk; ++ks) {
    const int cur = ks & 1, nxt = cur ^ 1;

    // kick off DMA of the next k-tile while we compute on the current one
    if (ks + 1 < nk) {
      const int kn = (ks + 1) << 5;
#if USE_TDM
      if (wave == 0) {
        tdm_load_tile((unsigned)(unsigned long long)&ldsA[nxt][0],
                      Ag + (size_t)kn * 2, K, 128);
        tdm_load_tile((unsigned)(unsigned long long)&ldsB[nxt][0],
                      Bg + (size_t)kn * 2, K, 128);
      }
#else
      stage_tile(&ldsA[nxt][0], A  + (size_t)m0 * K + kn, K, tid);
      stage_tile(&ldsB[nxt][0], WT + (size_t)n0 * K + kn, K, tid);
#endif
    }

    // fragments from LDS, then 16 WMMAs
    bf16x16 af[4], bfr[4];
#pragma unroll
    for (int i = 0; i < 4; ++i)
      af[i] = load_afrag(
          (const unsigned short*)&ldsA[cur][(rw + 16 * i + l15) * LDSROW + half * 16]);
#pragma unroll
    for (int j = 0; j < 4; ++j)
      bfr[j] = load_bfrag(
          (const unsigned short*)&ldsB[cur][(cw + 16 * j + l15) * LDSROW + half * 16]);

#pragma unroll
    for (int i = 0; i < 4; ++i)
#pragma unroll
      for (int j = 0; j < 4; ++j)
        acc[i][j] = __builtin_amdgcn_wmma_f32_16x16x32_bf16(
            false, af[i], false, bfr[j], (short)0, acc[i][j], false, false);

#if USE_TDM
    if (wave == 0 && ks + 1 < nk)
      __builtin_amdgcn_s_wait_tensorcnt((short)0);
#endif
    __syncthreads();
  }

  // epilogue: bias + relu
#pragma unroll
  for (int i = 0; i < 4; ++i)
#pragma unroll
    for (int j = 0; j < 4; ++j) {
      const int   col = n0 + cw + 16 * j + l15;
      const float bv  = bias[col];
#pragma unroll
      for (int r = 0; r < 8; ++r) {
        float v = fmaxf(acc[i][j][r] + bv, 0.f);
        const size_t row = (size_t)(m0 + rw + 16 * i + r + 8 * half);
        if (emit_bf16) outb[row * (size_t)ostride + ocol + col] = f2bf(v);
        else           outf[row * (size_t)ostride + ocol + col] = v;
      }
    }
}

// ---------------------------------------------------------------------------
// Flash attention: grid (S/64, H, B), 128 thr = 4 waves, wave = 16 q-rows.
// ---------------------------------------------------------------------------
__global__ __launch_bounds__(128)
void flash_kernel(const unsigned short* __restrict__ Q,
                  const unsigned short* __restrict__ Kb,
                  const unsigned short* __restrict__ VT,
                  float* __restrict__ Y)
{
  __shared__ unsigned short plds[4][16 * 64];   // per-wave P tile (bf16)

  const int lane = threadIdx.x & 31;
  const int wave = threadIdx.x >> 5;
  const int l15  = lane & 15;
  const int half = lane >> 4;
  const int b = blockIdx.z, h = blockIdx.y;
  const int q0 = blockIdx.x * 64 + wave * 16;
  const float L2E = 1.4426950408889634f;

  const unsigned short* qrow =
      Q + ((size_t)(b * S_ + q0 + l15)) * D_ + h * 64 + half * 16;
  const bf16x16 aq0 = load_afrag(qrow);
  const bf16x16 aq1 = load_afrag(qrow + 32);

  f32x8 o[4], mrow, lrow;
#pragma unroll
  for (int r = 0; r < 8; ++r) { mrow[r] = -3.0e38f; lrow[r] = 0.f; }
#pragma unroll
  for (int c = 0; c < 4; ++c)
#pragma unroll
    for (int r = 0; r < 8; ++r) o[c][r] = 0.f;

  const unsigned short* kbase = Kb + ((size_t)b * S_) * D_ + h * 64 + half * 16;
  const unsigned short* vbase = VT + ((size_t)(b * H_ + h)) * 64 * S_;

  for (int kb = 0; kb < S_; kb += 64) {
    f32x8 s[4];
#pragma unroll
    for (int c = 0; c < 4; ++c) {
      const unsigned short* kr = kbase + (size_t)(kb + c * 16 + l15) * D_;
      f32x8 z;
#pragma unroll
      for (int r = 0; r < 8; ++r) z[r] = 0.f;
      s[c] = __builtin_amdgcn_wmma_f32_16x16x32_bf16(
          false, aq0, false, load_bfrag(kr), (short)0, z, false, false);
      s[c] = __builtin_amdgcn_wmma_f32_16x16x32_bf16(
          false, aq1, false, load_bfrag(kr + 32), (short)0, s[c], false, false);
#pragma unroll
      for (int r = 0; r < 8; ++r) s[c][r] *= 0.125f;   // 1/sqrt(64)
    }

    f32x8 t;
#pragma unroll
    for (int r = 0; r < 8; ++r)
      t[r] = fmaxf(fmaxf(s[0][r], s[1][r]), fmaxf(s[2][r], s[3][r]));
#pragma unroll
    for (int off = 1; off < 16; off <<= 1)
#pragma unroll
      for (int r = 0; r < 8; ++r)
        t[r] = fmaxf(t[r], __shfl_xor(t[r], off, 32));

    f32x8 mnew, alpha;
#pragma unroll
    for (int r = 0; r < 8; ++r) {
      mnew[r]  = fmaxf(mrow[r], t[r]);
      alpha[r] = exp2f((mrow[r] - mnew[r]) * L2E);
      mrow[r]  = mnew[r];
    }

    f32x8 psum;
#pragma unroll
    for (int r = 0; r < 8; ++r) psum[r] = 0.f;
#pragma unroll
    for (int c = 0; c < 4; ++c)
#pragma unroll
      for (int r = 0; r < 8; ++r) {
        float pv = exp2f((s[c][r] - mnew[r]) * L2E);
        s[c][r] = pv;
        psum[r] += pv;
      }
#pragma unroll
    for (int off = 1; off < 16; off <<= 1)
#pragma unroll
      for (int r = 0; r < 8; ++r) psum[r] += __shfl_xor(psum[r], off, 32);
#pragma unroll
    for (int r = 0; r < 8; ++r) {
      lrow[r] = lrow[r] * alpha[r] + psum[r];
#pragma unroll
      for (int c = 0; c < 4; ++c) o[c][r] *= alpha[r];
    }

    // re-layout P (C-layout -> A-fragment) through wave-private LDS
#pragma unroll
    for (int c = 0; c < 4; ++c)
#pragma unroll
      for (int r = 0; r < 8; ++r)
        plds[wave][(r + 8 * half) * 64 + c * 16 + l15] = f2bf(s[c][r]);

    const unsigned short* pr =
        (const unsigned short*)&plds[wave][l15 * 64 + half * 16];
    const bf16x16 ap0 = load_afrag(pr);
    const bf16x16 ap1 = load_afrag(pr + 32);

#pragma unroll
    for (int c = 0; c < 4; ++c) {
      const unsigned short* vr =
          vbase + (size_t)(c * 16 + l15) * S_ + kb + half * 16;
      o[c] = __builtin_amdgcn_wmma_f32_16x16x32_bf16(
          false, ap0, false, load_bfrag(vr), (short)0, o[c], false, false);
      o[c] = __builtin_amdgcn_wmma_f32_16x16x32_bf16(
          false, ap1, false, load_bfrag(vr + 32), (short)0, o[c], false, false);
    }
  }

#pragma unroll
  for (int c = 0; c < 4; ++c)
#pragma unroll
    for (int r = 0; r < 8; ++r) {
      const size_t row = (size_t)(b * S_ + q0 + r + 8 * half);
      Y[row * D_ + h * 64 + c * 16 + l15] = o[c][r] / lrow[r];
    }
}

// ---------------------------------------------------------------------------
// Host-side orchestration
// ---------------------------------------------------------------------------
static inline size_t alignup256(size_t x) { return (x + 255) & ~(size_t)255; }

static void run_chain(const float* x, void* const* d_in, int pbase,
                      const int* widths, int nlayers,
                      float* feats, unsigned short* Abf, unsigned short* WTbuf,
                      float* outf, unsigned short* outb, hipStream_t stream)
{
  copyin_kernel<<<(M_ * D_) / 256, 256, 0, stream>>>(x, feats);
  int ocol = D_;
  for (int l = 0; l < nlayers; ++l) {
    const int fi = widths[l];
    const float* g    = (const float*)d_in[pbase + 4 * l + 0];
    const float* beta = (const float*)d_in[pbase + 4 * l + 1];
    const float* W    = (const float*)d_in[pbase + 4 * l + 2];
    const float* bias = (const float*)d_in[pbase + 4 * l + 3];

    wtconv_kernel<<<dim3(D_, (fi + 255) / 256), 256, 0, stream>>>(W, WTbuf, fi);
    featnorm_kernel<<<M_, 256, 0, stream>>>(feats, FSTRIDE, fi, g, beta, Abf);

    const bool last = (l == nlayers - 1);
    gemm_fn_kernel<<<dim3(M_ / 128, D_ / 128), 128, 0, stream>>>(
        Abf, WTbuf, bias,
        last ? outf : feats, outb,
        fi,
        last ? D_ : FSTRIDE,
        last ? 0 : ocol,
        (last && outb != nullptr) ? 1 : 0);
    ocol += D_;
  }
}

extern "C" void kernel_launch(void* const* d_in, const int* in_sizes, int n_in,
                              void* d_out, int out_size, void* d_ws,
                              size_t ws_size, hipStream_t stream)
{
  (void)in_sizes; (void)n_in; (void)out_size; (void)ws_size;

  char* p = (char*)d_ws;
  size_t off = 0;
  auto take = [&](size_t bytes) -> void* {
    void* r = p + off;
    off = alignup256(off + bytes);
    return r;
  };

  float*          feats = (float*)         take((size_t)M_ * FSTRIDE * 4);
  unsigned short* Abf   = (unsigned short*)take((size_t)M_ * MAXFI * 2);
  unsigned short* WTb   = (unsigned short*)take((size_t)MAXFI * D_ * 2);
  unsigned short* QBF   = (unsigned short*)take((size_t)M_ * D_ * 2);
  unsigned short* KBF   = (unsigned short*)take((size_t)M_ * D_ * 2);
  unsigned short* VBF   = (unsigned short*)take((size_t)M_ * D_ * 2);
  unsigned short* VTb   = (unsigned short*)take((size_t)M_ * D_ * 2);
  float*          Y     = (float*)         take((size_t)M_ * D_ * 4);

  static const int qkv_w[3] = {1024, 2048, 3072};
  static const int o_w[5]   = {1024, 2048, 3072, 4096, 5120};

  const float* q = (const float*)d_in[0];
  const float* k = (const float*)d_in[1];
  const float* v = (const float*)d_in[2];

  run_chain(q, d_in,  3, qkv_w, 3, feats, Abf, WTb, nullptr, QBF, stream);
  run_chain(k, d_in, 15, qkv_w, 3, feats, Abf, WTb, nullptr, KBF, stream);
  run_chain(v, d_in, 27, qkv_w, 3, feats, Abf, WTb, nullptr, VBF, stream);

  {
    size_t tot = (size_t)B_ * H_ * 64 * S_;
    vtrans_kernel<<<(unsigned)((tot + 255) / 256), 256, 0, stream>>>(VBF, VTb);
  }

  flash_kernel<<<dim3(S_ / 64, H_, B_), 128, 0, stream>>>(QBF, KBF, VTb, Y);

  run_chain(Y, d_in, 39, o_w, 5, feats, Abf, WTb, (float*)d_out, nullptr,
            stream);
}
